// BatchLSTMCRFTagger_89026082112119
// MI455X (gfx1250) — matrix operand order repaired and checked by reading, not compiled, not run on hardware
//
#include <hip/hip_runtime.h>
#include <hip/hip_bf16.h>
#include <math.h>

typedef __attribute__((ext_vector_type(16))) _Float16 v16h;
typedef __attribute__((ext_vector_type(8)))  float    v8f;

#define L_SEQ  256
#define BATCH  64
#define EMB    256
#define HID    256
#define G4H    1024
#define HDIM   512
#define NHEADS 4
#define DK     50
#define QKV    200
#define T_TAGS 32
#define LINF   2136   // 3 * 712
#define SEG    712

// gfx1250 async global->LDS copy path (guarded; falls back to sync staging)
#ifndef ATHENA_ASYNC_LDS
#  if defined(__has_builtin)
#    if __has_builtin(__builtin_amdgcn_global_load_async_to_lds_b32) && \
        __has_builtin(__builtin_amdgcn_s_wait_asynccnt)
#      define ATHENA_ASYNC_LDS 1
#    endif
#  endif
#endif

// ---------------------------------------------------------------------------
// WMMA fragment index helpers (CDNA5 ISA 7.12.2, wave32, 16x16x32 f16)
//  A (16xK): lane.m = lane&15 is the row; VGPR v holds K pair
//            k = (v<4 ? 2v : 16+2(v-4)) + (lane>=16 ? 8 : 0) + (j&1)
//            -> per lane two contiguous 16B runs (compiler emits b128 loads)
//  B (Kx16): lane.n = lane&15 is the column; k same as A (table order)
//  C/D:      element r -> (M = r + 8*(lane>=16), N = lane&15)
// ---------------------------------------------------------------------------
__device__ __forceinline__ int a_k_of(int j, int half) {
  int v = j >> 1;
  int base = (v < 4) ? (2 * v) : (16 + 2 * (v - 4));
  return base + (half ? 8 : 0) + (j & 1);
}
__device__ __forceinline__ int b_k_of(int j, int half) { return a_k_of(j, half); }

__device__ __forceinline__ v8f wmma_f16(v16h a, v16h b, v8f c) {
  return __builtin_amdgcn_wmma_f32_16x16x32_f16(false, a, false, b, (short)0, c, false, false);
}

__device__ __forceinline__ float sigmoidf_(float x) { return 1.0f / (1.0f + __expf(-x)); }

// A fragment from row-major f16 global (M x lda), tile rows row0..row0+15
__device__ __forceinline__ v16h load_a_g(const _Float16* __restrict__ A, int lda,
                                         int row0, int k0) {
  const int lane = threadIdx.x & 31, half = lane >> 4, m = lane & 15;
  v16h a;
#pragma unroll
  for (int j = 0; j < 16; ++j)
    a[j] = A[(size_t)(row0 + m) * lda + k0 + a_k_of(j, half)];
  return a;
}

// B fragment from weight stored (N x K) row-major (out = A @ W^T)
__device__ __forceinline__ v16h load_b_nk(const _Float16* __restrict__ W, int ldk,
                                          int n0, int k0) {
  const int lane = threadIdx.x & 31, half = lane >> 4, n = lane & 15;
  v16h b;
#pragma unroll
  for (int j = 0; j < 16; ++j)
    b[j] = W[(size_t)(n0 + n) * ldk + k0 + b_k_of(j, half)];
  return b;
}

// B fragment from weight stored (K x N) row-major with bounds guards
__device__ __forceinline__ v16h load_b_kn(const _Float16* __restrict__ W, int ldn,
                                          int n0, int k0, int kmax, int nmax) {
  const int lane = threadIdx.x & 31, half = lane >> 4;
  const int n = n0 + (lane & 15);
  v16h b;
#pragma unroll
  for (int j = 0; j < 16; ++j) {
    int k = k0 + b_k_of(j, half);
    b[j] = (k < kmax && n < nmax) ? W[(size_t)k * ldn + n] : (_Float16)0.0f;
  }
  return b;
}

// ---------------------------------------------------------------------------
// Utility kernels
// ---------------------------------------------------------------------------
__global__ void f32_to_f16_kernel(const float* __restrict__ s, _Float16* __restrict__ d, int n) {
  for (int i = blockIdx.x * blockDim.x + threadIdx.x; i < n; i += gridDim.x * blockDim.x)
    d[i] = (_Float16)s[i];
}

__global__ void embed_kernel(const int* __restrict__ sent, const float* __restrict__ emb,
                             _Float16* __restrict__ x0) {
  const int total = L_SEQ * BATCH * EMB;
  for (int i = blockIdx.x * blockDim.x + threadIdx.x; i < total; i += gridDim.x * blockDim.x) {
    int row = i >> 8, e = i & 255;
    x0[i] = (_Float16)emb[(size_t)sent[row] * EMB + e];
  }
}

// ---------------------------------------------------------------------------
// Persistent bidirectional LSTM layer.  grid = 2 (blockIdx 0 = fwd, 1 = rev),
// block = 1024 threads (32 waves).  Fused input+recurrent GEMM per step:
//   g = x[t] @ Wih^T + h @ Whh^T + b   (64 x 1024, K = Kin + 256)
// h lives in LDS (f16), cell state lives in accumulator VGPRs for the whole
// sequence.  Wave w owns quads {2w, 2w+1}: quad = (mtile, jtile), computes all
// four gate tiles for those 16 hidden columns.  Weights stream from L2 (all
// f16 weights ~6MB << 192MB L2); next step's x row is prefetched behind the
// current step's WMMA chain.
// ---------------------------------------------------------------------------
#define HSTR 264
__global__ __launch_bounds__(1024) void lstm_layer_kernel(
    const _Float16* __restrict__ xin, int Kin,
    const _Float16* __restrict__ Wih_f, const _Float16* __restrict__ Whh_f,
    const float* __restrict__ bias_f,
    const _Float16* __restrict__ Wih_r, const _Float16* __restrict__ Whh_r,
    const float* __restrict__ bias_r,
    _Float16* __restrict__ out_f16 /* (L,B,512), fwd->cols 0..255, rev->256..511 */) {
  __shared__ _Float16 hsh[BATCH * HSTR];

  const int rev = blockIdx.x;
  const _Float16* Wih  = rev ? Wih_r  : Wih_f;
  const _Float16* Whh  = rev ? Whh_r  : Whh_f;
  const float*    bptr = rev ? bias_r : bias_f;
  const int col_off = rev ? HID : 0;

  const int tid = threadIdx.x;
  const int wave = tid >> 5;
  const int lane = tid & 31, half = lane >> 4, mlane = lane & 15;

  for (int i = tid; i < BATCH * HSTR; i += 1024) hsh[i] = (_Float16)0.0f;

  v8f cst[2], acc[2][4];
#pragma unroll
  for (int q = 0; q < 2; ++q)
#pragma unroll
    for (int r = 0; r < 8; ++r) cst[q][r] = 0.0f;
  __syncthreads();

  const int xchunk = (BATCH * Kin) / 1024;   // f16 elems per thread of one x row

  for (int step = 0; step < L_SEQ; ++step) {
    const int tt = rev ? (L_SEQ - 1 - step) : step;
    const _Float16* xrow = xin + (size_t)tt * BATCH * Kin;

    // prefetch next step's x row (streamed once; hide behind WMMA chain)
    if (step + 1 < L_SEQ) {
      const int tn = rev ? (L_SEQ - 2 - step) : (step + 1);
      __builtin_prefetch(xin + (size_t)tn * BATCH * Kin + (size_t)tid * xchunk, 0, 0);
    }

#pragma unroll
    for (int q = 0; q < 2; ++q) {
      const int qid = wave * 2 + q;
      const int mtile = qid >> 4, jtile = qid & 15;
      const int nj = jtile * 16;
#pragma unroll
      for (int g = 0; g < 4; ++g) {
        float bv = bptr[g * HID + nj + mlane];
#pragma unroll
        for (int r = 0; r < 8; ++r) acc[q][g][r] = bv;
      }
      // input projection part:   x[t] (64 x Kin) @ Wih^T
      for (int k0 = 0; k0 < Kin; k0 += 32) {
        v16h a = load_a_g(xrow, Kin, mtile * 16, k0);
#pragma unroll
        for (int g = 0; g < 4; ++g)
          acc[q][g] = wmma_f16(a, load_b_nk(Wih, Kin, g * HID + nj, k0), acc[q][g]);
      }
      // recurrent part:          h (64 x 256, LDS) @ Whh^T
      for (int k0 = 0; k0 < HID; k0 += 32) {
        v16h a;
#pragma unroll
        for (int j = 0; j < 16; ++j)
          a[j] = hsh[(mtile * 16 + mlane) * HSTR + k0 + a_k_of(j, half)];
#pragma unroll
        for (int g = 0; g < 4; ++g)
          acc[q][g] = wmma_f16(a, load_b_nk(Whh, HID, g * HID + nj, k0), acc[q][g]);
      }
    }
    __syncthreads();   // everyone done reading hsh

#pragma unroll
    for (int q = 0; q < 2; ++q) {
      const int qid = wave * 2 + q;
      const int mtile = qid >> 4, jtile = qid & 15;
#pragma unroll
      for (int r = 0; r < 8; ++r) {
        float iv = sigmoidf_(acc[q][0][r]);
        float fv = sigmoidf_(acc[q][1][r]);
        float gv = tanhf(acc[q][2][r]);
        float ov = sigmoidf_(acc[q][3][r]);
        float cn = fv * cst[q][r] + iv * gv;
        float hv = ov * tanhf(cn);
        cst[q][r] = cn;
        int M = mtile * 16 + r + 8 * half;   // batch row
        int N = jtile * 16 + mlane;          // hidden col
        _Float16 hh = (_Float16)hv;
        hsh[M * HSTR + N] = hh;
        out_f16[((size_t)tt * BATCH + M) * HDIM + col_off + N] = hh;
      }
    }
    __syncthreads();   // new h visible for next step
  }
}

// ---------------------------------------------------------------------------
// q/k/v projection:  out_f16 (16384 x 200) = A (16384 x 512) @ W (512 x 200) + b
// grid (1024 mtiles, 13 ntiles), one wave per block
// ---------------------------------------------------------------------------
__global__ __launch_bounds__(32) void qkv_gemm_kernel(
    const _Float16* __restrict__ A, const _Float16* __restrict__ W,
    const float* __restrict__ bias, _Float16* __restrict__ out) {
  const int mtile = blockIdx.x, nt = blockIdx.y;
  const int lane = threadIdx.x & 31, half = lane >> 4, mlane = lane & 15;
  const int n = nt * 16 + mlane;
  float bv = (n < QKV) ? bias[n] : 0.0f;
  v8f acc;
#pragma unroll
  for (int r = 0; r < 8; ++r) acc[r] = bv;
  for (int k0 = 0; k0 < HDIM; k0 += 32) {
    v16h a = load_a_g(A, HDIM, mtile * 16, k0);
    v16h b = load_b_kn(W, QKV, nt * 16, k0, HDIM, QKV);
    acc = wmma_f16(a, b, acc);
  }
  if (n < QKV) {
#pragma unroll
    for (int r = 0; r < 8; ++r) {
      int row = mtile * 16 + r + 8 * half;
      out[(size_t)row * QKV + n] = (_Float16)acc[r];
    }
  }
}

// ---------------------------------------------------------------------------
// Flash-style attention.  One block per (b, head); K/V staged in LDS (d padded
// to 52) -- via GLOBAL_LOAD_ASYNC_TO_LDS_B32 when available (ASYNCcnt-tracked,
// no VGPR round trip), sync fallback otherwise.  Each of 4 waves owns 4 query
// tiles; online softmax with 32-key chunks; P goes through per-wave LDS
// scratch to re-enter WMMA as an A-fragment.
// ---------------------------------------------------------------------------
#define KVSTR 52
__global__ __launch_bounds__(128) void attn_kernel(
    const _Float16* __restrict__ qg, const _Float16* __restrict__ kg,
    const _Float16* __restrict__ vg, _Float16* __restrict__ og) {
  __shared__ _Float16 Ksh[L_SEQ * KVSTR];
  __shared__ _Float16 Vsh[L_SEQ * KVSTR];
  __shared__ _Float16 Psh[4 * 16 * 32];

  const int bh = blockIdx.x, bidx = bh >> 2, hd = bh & 3;
  const int tid = threadIdx.x, wv = tid >> 5;
  const int lane = tid & 31, half = lane >> 4, mlane = lane & 15;

#ifdef ATHENA_ASYNC_LDS
  // zero the two pad columns (plain DS stores; barrier covers them)
  for (int m = tid; m < L_SEQ; m += 128) {
    Ksh[m * KVSTR + DK] = (_Float16)0.0f;  Ksh[m * KVSTR + DK + 1] = (_Float16)0.0f;
    Vsh[m * KVSTR + DK] = (_Float16)0.0f;  Vsh[m * KVSTR + DK + 1] = (_Float16)0.0f;
  }
  // async copy 25 dwords per row (50 f16, 4B-aligned: head offset = hd*100 B)
  for (int i = tid; i < L_SEQ * (DK / 2); i += 128) {
    int m = i / (DK / 2), dw = i - m * (DK / 2);
    size_t goff = ((size_t)(m * BATCH + bidx)) * QKV + hd * DK + dw * 2;
    int loff = m * KVSTR + dw * 2;
    __builtin_amdgcn_global_load_async_to_lds_b32(
        (int*)(kg + goff), (int*)(Ksh + loff), 0, 0);
    __builtin_amdgcn_global_load_async_to_lds_b32(
        (int*)(vg + goff), (int*)(Vsh + loff), 0, 0);
  }
  __builtin_amdgcn_s_wait_asynccnt(0);
#else
  for (int i = tid; i < L_SEQ * KVSTR; i += 128) {
    int m = i / KVSTR, d = i - m * KVSTR;
    _Float16 kvv = (_Float16)0.0f, vvv = (_Float16)0.0f;
    if (d < DK) {
      size_t src = ((size_t)(m * BATCH + bidx)) * QKV + hd * DK + d;
      kvv = kg[src];
      vvv = vg[src];
    }
    Ksh[i] = kvv;
    Vsh[i] = vvv;
  }
#endif
  __syncthreads();

  const float scale = rsqrtf((float)DK);
  for (int qt = wv; qt < 16; qt += 4) {
    v16h qa[2];
#pragma unroll
    for (int c = 0; c < 2; ++c) {
#pragma unroll
      for (int j = 0; j < 16; ++j) {
        int d = c * 32 + a_k_of(j, half);
        int l = qt * 16 + mlane;
        qa[c][j] = (d < DK) ? qg[((size_t)(l * BATCH + bidx)) * QKV + hd * DK + d]
                            : (_Float16)0.0f;
      }
    }
    float rm[8], rl[8];
    v8f O[4];
#pragma unroll
    for (int r = 0; r < 8; ++r) { rm[r] = -1e30f; rl[r] = 0.0f; }
#pragma unroll
    for (int nt = 0; nt < 4; ++nt)
#pragma unroll
      for (int r = 0; r < 8; ++r) O[nt][r] = 0.0f;

    for (int kc = 0; kc < L_SEQ; kc += 32) {
      v8f S0, S1;
#pragma unroll
      for (int r = 0; r < 8; ++r) { S0[r] = 0.0f; S1[r] = 0.0f; }
#pragma unroll
      for (int c = 0; c < 2; ++c) {       // K-dim (d) chunks 0 and 32
        v16h b0, b1;
#pragma unroll
        for (int j = 0; j < 16; ++j) {
          int d = c * 32 + b_k_of(j, half);
          b0[j] = (d < DK) ? Ksh[(kc + mlane) * KVSTR + d] : (_Float16)0.0f;
          b1[j] = (d < DK) ? Ksh[(kc + 16 + mlane) * KVSTR + d] : (_Float16)0.0f;
        }
        S0 = wmma_f16(qa[c], b0, S0);
        S1 = wmma_f16(qa[c], b1, S1);
      }
      // online softmax update (row r lives in the 16 lanes of this half-wave)
#pragma unroll
      for (int r = 0; r < 8; ++r) {
        float s0 = S0[r] * scale, s1 = S1[r] * scale;
        float cm = fmaxf(s0, s1);
        for (int off = 1; off < 16; off <<= 1) cm = fmaxf(cm, __shfl_xor(cm, off, 32));
        float mnew = fmaxf(rm[r], cm);
        float corr = __expf(rm[r] - mnew);
        float p0 = __expf(s0 - mnew), p1 = __expf(s1 - mnew);
        float rs = p0 + p1;
        for (int off = 1; off < 16; off <<= 1) rs += __shfl_xor(rs, off, 32);
        rl[r] = rl[r] * corr + rs;
        rm[r] = mnew;
#pragma unroll
        for (int nt = 0; nt < 4; ++nt) O[nt][r] *= corr;
        int row = r + 8 * half;
        Psh[wv * 512 + row * 32 + mlane]      = (_Float16)p0;
        Psh[wv * 512 + row * 32 + 16 + mlane] = (_Float16)p1;
      }
      // O += P (16x32) @ V (32x64)
      v16h pa;
#pragma unroll
      for (int j = 0; j < 16; ++j)
        pa[j] = Psh[wv * 512 + mlane * 32 + a_k_of(j, half)];
#pragma unroll
      for (int nt = 0; nt < 4; ++nt) {
        v16h vb;
        int d = nt * 16 + mlane;
#pragma unroll
        for (int j = 0; j < 16; ++j) {
          int kk = kc + b_k_of(j, half);
          vb[j] = (d < DK) ? Vsh[kk * KVSTR + d] : (_Float16)0.0f;
        }
        O[nt] = wmma_f16(pa, vb, O[nt]);
      }
    }
#pragma unroll
    for (int r = 0; r < 8; ++r) {
      float inv = 1.0f / rl[r];
      int l = qt * 16 + r + 8 * half;
#pragma unroll
      for (int nt = 0; nt < 4; ++nt) {
        int d = nt * 16 + mlane;
        if (d < DK)
          og[((size_t)(l * BATCH + bidx)) * QKV + hd * DK + d] = (_Float16)(O[nt][r] * inv);
      }
    }
  }
}

// ---------------------------------------------------------------------------
// Window-concat + final linear:
//   feats (L*B, 32) = [hcat[t+1], hcat[t], hcat[t-1]] @ W2t + b2t
// hcat = concat(h1 (512), attn_out (200)) read on the fly (no 70MB A matrix).
// grid (1024 mtiles, 2 ntiles), one wave per block.
// ---------------------------------------------------------------------------
__global__ __launch_bounds__(32) void window_linear_kernel(
    const _Float16* __restrict__ h1, const _Float16* __restrict__ og,
    const _Float16* __restrict__ W2t, const float* __restrict__ b2t,
    float* __restrict__ feats) {
  const int mtile = blockIdx.x, nt = blockIdx.y;
  const int lane = threadIdx.x & 31, half = lane >> 4, mlane = lane & 15;
  const int n = nt * 16 + mlane;
  v8f acc;
  float bv = b2t[n];
#pragma unroll
  for (int r = 0; r < 8; ++r) acc[r] = bv;

  const int row = mtile * 16 + mlane;     // A-fragment row for this lane
  const int t = row >> 6, bb = row & 63;  // B = 64

  for (int k0 = 0; k0 < LINF; k0 += 32) {
    v16h a;
#pragma unroll
    for (int j = 0; j < 16; ++j) {
      int k = k0 + a_k_of(j, half);
      _Float16 val = (_Float16)0.0f;
      if (k < LINF) {
        int seg = k / SEG, off = k - seg * SEG;
        int tp = t + 1 - seg;
        if (tp >= 0 && tp < L_SEQ) {
          size_t rb = (size_t)(tp * BATCH + bb);
          val = (off < HDIM) ? h1[rb * HDIM + off] : og[rb * QKV + (off - HDIM)];
        }
      }
      a[j] = val;
    }
    v16h bfr;
#pragma unroll
    for (int j = 0; j < 16; ++j) {
      int k = k0 + b_k_of(j, half);
      bfr[j] = (k < LINF) ? W2t[(size_t)k * T_TAGS + n] : (_Float16)0.0f;
    }
    acc = wmma_f16(a, bfr, acc);
  }
#pragma unroll
  for (int r = 0; r < 8; ++r) {
    int orow = mtile * 16 + r + 8 * half;
    feats[(size_t)orow * T_TAGS + n] = acc[r];
  }
}

// ---------------------------------------------------------------------------
// Viterbi max-plus DP + backtrack.  One block; dp (64x32) double-buffered in
// LDS; backpointers in workspace; scores + paths written to d_out as floats.
// ---------------------------------------------------------------------------
__global__ __launch_bounds__(256) void viterbi_kernel(
    const float* __restrict__ feats, const float* __restrict__ trans,
    const float* __restrict__ start_t, const float* __restrict__ stop_t,
    int* __restrict__ bp, float* __restrict__ out) {
  __shared__ float dp[2][BATCH * T_TAGS];
  __shared__ float trsh[T_TAGS * T_TAGS];
  __shared__ int last_sh[BATCH];
  const int tid = threadIdx.x;

  for (int i = tid; i < T_TAGS * T_TAGS; i += 256) trsh[i] = trans[i];
  for (int i = tid; i < BATCH * T_TAGS; i += 256)
    dp[0][i] = feats[i] + start_t[i & 31];
  __syncthreads();

  int cur = 0;
  for (int t = 1; t < L_SEQ; ++t) {
    for (int p = tid; p < BATCH * T_TAGS; p += 256) {
      int b = p >> 5, n = p & 31;
      float best = -1e30f; int arg = 0;
      for (int pv = 0; pv < T_TAGS; ++pv) {
        float s = dp[cur][b * T_TAGS + pv] + trsh[n * T_TAGS + pv];
        if (s > best) { best = s; arg = pv; }
      }
      dp[1 - cur][p] = best + feats[(size_t)t * BATCH * T_TAGS + p];
      bp[((size_t)(t - 1) * BATCH + b) * T_TAGS + n] = arg;
    }
    __syncthreads();
    cur = 1 - cur;
  }
  if (tid < BATCH) {
    float best = -1e30f; int arg = 0;
    for (int n = 0; n < T_TAGS; ++n) {
      float s = dp[cur][tid * T_TAGS + n] + stop_t[n];
      if (s > best) { best = s; arg = n; }
    }
    out[tid] = best;
    last_sh[tid] = arg;
  }
  __syncthreads();
  if (tid < BATCH) {
    int tag = last_sh[tid];
    out[64 + (size_t)(L_SEQ - 1) * BATCH + tid] = (float)tag;
    for (int t = L_SEQ - 2; t >= 0; --t) {
      tag = bp[((size_t)t * BATCH + tid) * T_TAGS + tag];
      out[64 + (size_t)t * BATCH + tid] = (float)tag;
    }
  }
}

// ---------------------------------------------------------------------------
// Host side
// ---------------------------------------------------------------------------
extern "C" void kernel_launch(void* const* d_in, const int* in_sizes, int n_in,
                              void* d_out, int out_size, void* d_ws, size_t ws_size,
                              hipStream_t stream) {
  const int*   sent    = (const int*)d_in[0];
  const float* emb     = (const float*)d_in[1];
  const float* Wih_l0f = (const float*)d_in[2];
  const float* Whh_l0f = (const float*)d_in[3];
  const float* b_l0f   = (const float*)d_in[4];
  const float* Wih_l0r = (const float*)d_in[5];
  const float* Whh_l0r = (const float*)d_in[6];
  const float* b_l0r   = (const float*)d_in[7];
  const float* Wih_l1f = (const float*)d_in[8];
  const float* Whh_l1f = (const float*)d_in[9];
  const float* b_l1f   = (const float*)d_in[10];
  const float* Wih_l1r = (const float*)d_in[11];
  const float* Whh_l1r = (const float*)d_in[12];
  const float* b_l1r   = (const float*)d_in[13];
  const float* Wq      = (const float*)d_in[14];
  const float* bq      = (const float*)d_in[15];
  const float* Wk      = (const float*)d_in[16];
  const float* bk      = (const float*)d_in[17];
  const float* Wv      = (const float*)d_in[18];
  const float* bv      = (const float*)d_in[19];
  const float* W2t     = (const float*)d_in[20];
  const float* b2t     = (const float*)d_in[21];
  const float* trans   = (const float*)d_in[22];
  const float* start_t = (const float*)d_in[23];
  const float* stop_t  = (const float*)d_in[24];
  float* out = (float*)d_out;

  char* ws = (char*)d_ws;
  size_t ofs = 0;
  auto alloc = [&](size_t bytes) -> char* {
    char* p = ws + ofs;
    ofs += (bytes + 255) & ~(size_t)255;
    return p;
  };
  const size_t ROWS = (size_t)L_SEQ * BATCH;   // 16384

  _Float16* x0   = (_Float16*)alloc(ROWS * EMB  * 2);
  _Float16* h0   = (_Float16*)alloc(ROWS * HDIM * 2);
  _Float16* h1   = (_Float16*)alloc(ROWS * HDIM * 2);
  _Float16* qb   = (_Float16*)alloc(ROWS * QKV  * 2);
  _Float16* kb_  = (_Float16*)alloc(ROWS * QKV  * 2);
  _Float16* vb_  = (_Float16*)alloc(ROWS * QKV  * 2);
  _Float16* ob   = (_Float16*)alloc(ROWS * QKV  * 2);
  float*    feats = (float*)  alloc(ROWS * T_TAGS * 4);
  int*      bp    = (int*)    alloc((size_t)(L_SEQ - 1) * BATCH * T_TAGS * 4);

  _Float16* wih0f = (_Float16*)alloc((size_t)G4H * EMB  * 2);
  _Float16* whh0f = (_Float16*)alloc((size_t)G4H * HID  * 2);
  _Float16* wih0r = (_Float16*)alloc((size_t)G4H * EMB  * 2);
  _Float16* whh0r = (_Float16*)alloc((size_t)G4H * HID  * 2);
  _Float16* wih1f = (_Float16*)alloc((size_t)G4H * HDIM * 2);
  _Float16* whh1f = (_Float16*)alloc((size_t)G4H * HID  * 2);
  _Float16* wih1r = (_Float16*)alloc((size_t)G4H * HDIM * 2);
  _Float16* whh1r = (_Float16*)alloc((size_t)G4H * HID  * 2);
  _Float16* wqh   = (_Float16*)alloc((size_t)HDIM * QKV * 2);
  _Float16* wkh   = (_Float16*)alloc((size_t)HDIM * QKV * 2);
  _Float16* wvh   = (_Float16*)alloc((size_t)HDIM * QKV * 2);
  _Float16* w2th  = (_Float16*)alloc((size_t)LINF * T_TAGS * 2);

  auto cvt = [&](const float* s, _Float16* d, int n) {
    f32_to_f16_kernel<<<(n + 255) / 256, 256, 0, stream>>>(s, d, n);
  };
  cvt(Wih_l0f, wih0f, G4H * EMB);  cvt(Whh_l0f, whh0f, G4H * HID);
  cvt(Wih_l0r, wih0r, G4H * EMB);  cvt(Whh_l0r, whh0r, G4H * HID);
  cvt(Wih_l1f, wih1f, G4H * HDIM); cvt(Whh_l1f, whh1f, G4H * HID);
  cvt(Wih_l1r, wih1r, G4H * HDIM); cvt(Whh_l1r, whh1r, G4H * HID);
  cvt(Wq, wqh, HDIM * QKV); cvt(Wk, wkh, HDIM * QKV); cvt(Wv, wvh, HDIM * QKV);
  cvt(W2t, w2th, LINF * T_TAGS);

  embed_kernel<<<2048, 256, 0, stream>>>(sent, emb, x0);

  lstm_layer_kernel<<<2, 1024, 0, stream>>>(x0, EMB,
      wih0f, whh0f, b_l0f, wih0r, whh0r, b_l0r, h0);
  lstm_layer_kernel<<<2, 1024, 0, stream>>>(h0, HDIM,
      wih1f, whh1f, b_l1f, wih1r, whh1r, b_l1r, h1);

  dim3 gq(L_SEQ * BATCH / 16, (QKV + 15) / 16);
  qkv_gemm_kernel<<<gq, 32, 0, stream>>>(h1, wqh, bq, qb);
  qkv_gemm_kernel<<<gq, 32, 0, stream>>>(h1, wkh, bk, kb_);
  qkv_gemm_kernel<<<gq, 32, 0, stream>>>(h1, wvh, bv, vb_);

  attn_kernel<<<BATCH * NHEADS, 128, 0, stream>>>(qb, kb_, vb_, ob);

  dim3 gw(L_SEQ * BATCH / 16, T_TAGS / 16);
  window_linear_kernel<<<gw, 32, 0, stream>>>(h1, ob, w2th, b2t, feats);

  viterbi_kernel<<<1, 256, 0, stream>>>(feats, trans, start_t, stop_t, bp, out);
}
